// RNN_83906481094672
// MI455X (gfx1250) — matrix operand order repaired and checked by reading, not compiled, not run on hardware
//
#include <hip/hip_runtime.h>
#include <hip/hip_bf16.h>

// MI455X / gfx1250, wave32. Matrix ops via v_wmma_f32_16x16x32_bf16;
// LDS staging via global_load_async_to_lds_b128 (ASYNCcnt).

typedef __attribute__((ext_vector_type(16))) __bf16 v16bf;
typedef __attribute__((ext_vector_type(8)))  float  v8f;

#define B_ 64
#define S_ 512
#define I_ 1024
#define H_ 1024
#define O_ 1024

union Frag { uint4 u[2]; v16bf v; };

// f32 -> bf16 round-to-nearest-even
__device__ inline unsigned short f32_bf16_bits(float f) {
  unsigned u = __builtin_bit_cast(unsigned, f);
  return (unsigned short)((u + 0x7FFFu + ((u >> 16) & 1u)) >> 16);
}
__device__ inline __bf16 f32_to_bf16(float f) {
  unsigned short s = f32_bf16_bits(f);
  return __builtin_bit_cast(__bf16, s);
}

// A fragment (16x32 bf16, MxK). lane<16 -> M=lane, K={kb..kb+7, kb+16..kb+23};
// lane>=16 -> M=lane-16, K={kb+8..kb+15, kb+24..kb+31}. Two 16B loads.
__device__ inline v16bf load_a_frag(const __bf16* base, int ld, int lane) {
  const int m  = lane & 15;
  const int kh = (lane >> 4) << 3;
  const __bf16* p = base + (size_t)m * ld + kh;
  Frag f;
  f.u[0] = *(const uint4*)(p);
  f.u[1] = *(const uint4*)(p + 16);
  return f.v;
}

// B fragment (32x16 bf16, KxN); column n == row n of [N,K] weights.
// lane<16 -> N=lane, K={kb..kb+7, kb+16..kb+23}; lane>=16 -> K={kb+8..kb+15, kb+24..kb+31}.
__device__ inline v16bf load_b_frag(const __bf16* base, int ld, int lane) {
  const int n  = lane & 15;
  const int kh = (lane >> 4) << 3;
  const __bf16* p = base + (size_t)n * ld + kh;
  Frag f;
  f.u[0] = *(const uint4*)(p);
  f.u[1] = *(const uint4*)(p + 16);
  return f.v;
}

#define WMMA_BF16(A, Bf, C) \
  __builtin_amdgcn_wmma_f32_16x16x32_bf16(false, (A), false, (Bf), (short)0, (C), false, false)

// 32x64 output tile per wave, K=1024, software-pipelined (double-buffered frags).
__device__ inline void mm_tile_32x64(const __bf16* __restrict__ A0,
                                     const __bf16* __restrict__ A1,
                                     const __bf16* __restrict__ W,
                                     int K, int lane, v8f acc[2][4])
{
  v16bf a0 = load_a_frag(A0, K, lane);
  v16bf a1 = load_a_frag(A1, K, lane);
  v16bf b0 = load_b_frag(W + (size_t)0 * 16 * K, K, lane);
  v16bf b1 = load_b_frag(W + (size_t)1 * 16 * K, K, lane);
  v16bf b2 = load_b_frag(W + (size_t)2 * 16 * K, K, lane);
  v16bf b3 = load_b_frag(W + (size_t)3 * 16 * K, K, lane);
  for (int kb = 32; kb < K; kb += 32) {
    v16bf na0 = load_a_frag(A0 + kb, K, lane);
    v16bf na1 = load_a_frag(A1 + kb, K, lane);
    v16bf nb0 = load_b_frag(W + (size_t)0 * 16 * K + kb, K, lane);
    v16bf nb1 = load_b_frag(W + (size_t)1 * 16 * K + kb, K, lane);
    v16bf nb2 = load_b_frag(W + (size_t)2 * 16 * K + kb, K, lane);
    v16bf nb3 = load_b_frag(W + (size_t)3 * 16 * K + kb, K, lane);
    acc[0][0] = WMMA_BF16(a0, b0, acc[0][0]);
    acc[0][1] = WMMA_BF16(a0, b1, acc[0][1]);
    acc[0][2] = WMMA_BF16(a0, b2, acc[0][2]);
    acc[0][3] = WMMA_BF16(a0, b3, acc[0][3]);
    acc[1][0] = WMMA_BF16(a1, b0, acc[1][0]);
    acc[1][1] = WMMA_BF16(a1, b1, acc[1][1]);
    acc[1][2] = WMMA_BF16(a1, b2, acc[1][2]);
    acc[1][3] = WMMA_BF16(a1, b3, acc[1][3]);
    a0 = na0; a1 = na1; b0 = nb0; b1 = nb1; b2 = nb2; b3 = nb3;
  }
  acc[0][0] = WMMA_BF16(a0, b0, acc[0][0]);
  acc[0][1] = WMMA_BF16(a0, b1, acc[0][1]);
  acc[0][2] = WMMA_BF16(a0, b2, acc[0][2]);
  acc[0][3] = WMMA_BF16(a0, b3, acc[0][3]);
  acc[1][0] = WMMA_BF16(a1, b0, acc[1][0]);
  acc[1][1] = WMMA_BF16(a1, b1, acc[1][1]);
  acc[1][2] = WMMA_BF16(a1, b2, acc[1][2]);
  acc[1][3] = WMMA_BF16(a1, b3, acc[1][3]);
}

// ---------------- elementwise f32 -> bf16 convert (8 elems / thread) ----------
__global__ void cvt_f32_bf16(const float* __restrict__ in, __bf16* __restrict__ out, int n8) {
  int i = blockIdx.x * blockDim.x + threadIdx.x;
  if (i >= n8) return;
  const float4* p = (const float4*)in + (size_t)i * 2;
  float4 a = p[0], c = p[1];
  union { unsigned short s[8]; uint4 u; } r;
  r.s[0] = f32_bf16_bits(a.x); r.s[1] = f32_bf16_bits(a.y);
  r.s[2] = f32_bf16_bits(a.z); r.s[3] = f32_bf16_bits(a.w);
  r.s[4] = f32_bf16_bits(c.x); r.s[5] = f32_bf16_bits(c.y);
  r.s[6] = f32_bf16_bits(c.z); r.s[7] = f32_bf16_bits(c.w);
  *((uint4*)out + i) = r.u;
}

// ---------------- GEMM1: xproj[s,b,h] = x[b,s,:]@Wx[h,:] + bx[h] + bh[h] ------
__global__ __launch_bounds__(128) void gemm_xproj(
    const __bf16* __restrict__ xb,   // [B*S, I] bf16 (row = b*S + s)
    const __bf16* __restrict__ Wxb,  // [H, I]   bf16
    const float* __restrict__ bx, const float* __restrict__ bh,
    float* __restrict__ xproj)       // [S, B, H] f32
{
  const int lane  = threadIdx.x & 31;
  const int wave  = threadIdx.x >> 5;
  const int mt    = blockIdx.x;
  const int nbase = blockIdx.y * 256 + wave * 64;

  const __bf16* A0 = xb + (size_t)mt * 32 * I_;
  const __bf16* A1 = A0 + (size_t)16 * I_;
  v8f acc[2][4] = {};
  mm_tile_32x64(A0, A1, Wxb + (size_t)nbase * I_, I_, lane, acc);

  const int mloc = (lane >> 4) << 3;
#pragma unroll
  for (int i = 0; i < 2; ++i) {
#pragma unroll
    for (int j = 0; j < 4; ++j) {
      const int n = nbase + j * 16 + (lane & 15);
      const float bias = bx[n] + bh[n];
#pragma unroll
      for (int v = 0; v < 8; ++v) {
        const int r = mt * 32 + i * 16 + mloc + v;   // row = b*512 + s
        const int b = r >> 9, s = r & 511;
        xproj[((size_t)s * B_ + b) * H_ + n] = acc[i][j][v] + bias;
      }
    }
  }
}

// ---------------- persistent recurrence: h = tanh(xp_t + h @ Wh^T) -----------
// 64 WGs x 128 threads; WG g owns columns [16g,16g+16), Wh slice pinned in LDS.
__global__ __launch_bounds__(128) void rnn_steps(
    const __bf16* __restrict__ Whb,    // [H, H] bf16
    const float*  __restrict__ xproj,  // [S, B, H] f32
    __bf16* __restrict__ hbuf,         // [2, B, H] bf16 (buf0 = h0 = 0)
    unsigned* __restrict__ counter)
{
  __shared__ __bf16 ldsW[16 * 1032];   // padded stride: conflict-free b128 reads
  const int lane  = threadIdx.x & 31;
  const int wave  = threadIdx.x >> 5;  // wave w -> rows [16w, 16w+16)
  const int nbase = blockIdx.x * 16;

  // Async DMA the Wh slice (16 rows x 1024) straight into LDS (ASYNCcnt path):
  // each thread moves 128 halfs as 16B global->LDS async transfers.
  {
    const int row = (threadIdx.x * 128) >> 10;
    const int k0  = (threadIdx.x * 128) & 1023;
    const __bf16* src = Whb + (size_t)(nbase + row) * H_ + k0;
    __bf16*       dst = ldsW + row * 1032 + k0;
#pragma unroll
    for (int e = 0; e < 128; e += 8) {
      unsigned    lo = (unsigned)(uintptr_t)(dst + e);   // LDS byte offset (addr[31:0])
      const void* ga = (const void*)(src + e);
      asm volatile("global_load_async_to_lds_b128 %0, %1, off"
                   :: "v"(lo), "v"(ga) : "memory");
    }
    asm volatile("s_wait_asynccnt 0x0" ::: "memory");
  }
  __syncthreads();

  const int n    = nbase + (lane & 15);
  const int mloc = wave * 16 + ((lane >> 4) << 3);

  for (int t = 0; t < S_; ++t) {
    const __bf16* hsrc = hbuf + (size_t)(t & 1) * (B_ * H_);
    __bf16*       hdst = hbuf + (size_t)((t + 1) & 1) * (B_ * H_);
    const __bf16* Arow = hsrc + (size_t)(wave * 16) * H_;

    // hoist this step's xproj loads; prefetch next step's tile into cache
    const float* xp = xproj + (size_t)t * B_ * H_;
    float xpv[8];
#pragma unroll
    for (int v = 0; v < 8; ++v)
      xpv[v] = xp[(size_t)(mloc + v) * H_ + n];
    if (t + 1 < S_)
      __builtin_prefetch(xp + (size_t)B_ * H_ + (size_t)mloc * H_ + n, 0, 1);

    v8f acc = {};
    v16bf a = load_a_frag(Arow, H_, lane);
    v16bf b = load_b_frag(ldsW, 1032, lane);
    for (int kb = 32; kb < H_; kb += 32) {
      v16bf na = load_a_frag(Arow + kb, H_, lane);
      v16bf nb = load_b_frag(ldsW + kb, 1032, lane);
      acc = WMMA_BF16(a, b, acc);
      a = na; b = nb;
    }
    acc = WMMA_BF16(a, b, acc);

#pragma unroll
    for (int v = 0; v < 8; ++v) {
      float val = tanhf(acc[v] + xpv[v]);
      hdst[(size_t)(mloc + v) * H_ + n] = f32_to_bf16(val);
    }

    // grid-wide barrier per timestep
    __threadfence();
    __syncthreads();
    if (threadIdx.x == 0) {
      __hip_atomic_fetch_add(counter, 1u, __ATOMIC_ACQ_REL, __HIP_MEMORY_SCOPE_AGENT);
      const unsigned target = (unsigned)(t + 1) * gridDim.x;
      while (__hip_atomic_load(counter, __ATOMIC_ACQUIRE, __HIP_MEMORY_SCOPE_AGENT) < target)
        __builtin_amdgcn_s_sleep(1);
    }
    __syncthreads();
  }
}

// ---------------- GEMM3: out = h_last @ Wo^T + bo ----------------------------
__global__ __launch_bounds__(128) void gemm_out(
    const __bf16* __restrict__ hlast,  // [B, H] bf16 (hbuf buffer 0)
    const __bf16* __restrict__ Wob,    // [O, H] bf16
    const float* __restrict__ bo,
    float* __restrict__ out)           // [B, O] f32
{
  const int lane  = threadIdx.x & 31;
  const int wave  = threadIdx.x >> 5;
  const int mt    = blockIdx.x;
  const int nbase = blockIdx.y * 256 + wave * 64;

  const __bf16* A0 = hlast + (size_t)mt * 32 * H_;
  const __bf16* A1 = A0 + (size_t)16 * H_;
  v8f acc[2][4] = {};
  mm_tile_32x64(A0, A1, Wob + (size_t)nbase * H_, H_, lane, acc);

  const int mloc = (lane >> 4) << 3;
#pragma unroll
  for (int i = 0; i < 2; ++i) {
#pragma unroll
    for (int j = 0; j < 4; ++j) {
      const int n = nbase + j * 16 + (lane & 15);
      const float bias = bo[n];
#pragma unroll
      for (int v = 0; v < 8; ++v)
        out[(size_t)(mt * 32 + i * 16 + mloc + v) * O_ + n] = acc[i][j][v] + bias;
    }
  }
}

extern "C" void kernel_launch(void* const* d_in, const int* in_sizes, int n_in,
                              void* d_out, int out_size, void* d_ws, size_t ws_size,
                              hipStream_t stream) {
  (void)in_sizes; (void)n_in; (void)out_size; (void)ws_size;
  const float* x  = (const float*)d_in[0];
  const float* Wx = (const float*)d_in[1];
  const float* bx = (const float*)d_in[2];
  const float* Wh = (const float*)d_in[3];
  const float* bh = (const float*)d_in[4];
  const float* Wo = (const float*)d_in[5];
  const float* bo = (const float*)d_in[6];
  float* out = (float*)d_out;

  char* ws = (char*)d_ws;
  size_t off = 0;
  auto take = [&](size_t bytes) -> char* {
    char* p = ws + off;
    off += (bytes + 255) & ~(size_t)255;
    return p;
  };
  __bf16* xb    = (__bf16*)take((size_t)B_ * S_ * I_ * 2);  //  64 MB
  __bf16* Wxb   = (__bf16*)take((size_t)H_ * I_ * 2);       //   2 MB
  __bf16* Whb   = (__bf16*)take((size_t)H_ * H_ * 2);       //   2 MB
  __bf16* Wob   = (__bf16*)take((size_t)O_ * H_ * 2);       //   2 MB
  float*  xproj = (float*) take((size_t)S_ * B_ * H_ * 4);  // 128 MB
  __bf16* hbuf  = (__bf16*)take((size_t)2 * B_ * H_ * 2);   // 256 KB
  unsigned* counter = (unsigned*)take(256);

  hipMemsetAsync(counter, 0, 256, stream);
  hipMemsetAsync(hbuf, 0, (size_t)B_ * H_ * 2, stream);     // h0 = 0 (buffer 0)

  int n8 = (B_ * S_ * I_) / 8;
  cvt_f32_bf16<<<(n8 + 255) / 256, 256, 0, stream>>>(x, xb, n8);
  n8 = (H_ * I_) / 8;
  cvt_f32_bf16<<<(n8 + 255) / 256, 256, 0, stream>>>(Wx, Wxb, n8);
  cvt_f32_bf16<<<(n8 + 255) / 256, 256, 0, stream>>>(Wh, Whb, n8);
  cvt_f32_bf16<<<(n8 + 255) / 256, 256, 0, stream>>>(Wo, Wob, n8);

  gemm_xproj<<<dim3((B_ * S_) / 32, H_ / 256), 128, 0, stream>>>(xb, Wxb, bx, bh, xproj);
  rnn_steps<<<H_ / 16, 128, 0, stream>>>(Whb, xproj, hbuf, counter);
  // 512 steps: src/dst alternate, final h lands in buffer 0
  gemm_out<<<dim3(B_ / 32, O_ / 256), 128, 0, stream>>>(hbuf, Wob, bo, out);
}
